// MultiHeadAttention_19748259627056
// MI455X (gfx1250) — matrix operand order repaired and checked by reading, not compiled, not run on hardware
//
#include <hip/hip_runtime.h>

// ---------------------------------------------------------------------------
// Types for CDNA5 WMMA (wave32): v_wmma_f32_16x16x32_bf16
// ---------------------------------------------------------------------------
typedef __bf16 bf16_t;
typedef __attribute__((ext_vector_type(8)))  __bf16 bf16x8;
typedef __attribute__((ext_vector_type(16))) __bf16 bf16x16;
typedef __attribute__((ext_vector_type(8)))  float  floatx8;

#define B_   4
#define T_   2048
#define D_   1024
#define H_   16
#define HD_  64
#define M_   (B_ * T_)          // 8192 total rows
#define SROW 2056               // padded LDS row stride (floats), 32B aligned

__device__ __forceinline__ bf16_t to_bf16(float f) {
  union { float f; unsigned u; } v; v.f = f;
  unsigned r = v.u + 0x7FFFu + ((v.u >> 16) & 1u);   // round-to-nearest-even
  union { unsigned short s; bf16_t b; } o;
  o.s = (unsigned short)(r >> 16);
  return o.b;
}

__device__ __forceinline__ floatx8 wmma_bf16(bf16x16 a, bf16x16 b, floatx8 c) {
  // (neg_a, A, neg_b, B, c_mod, C, reuse_a, reuse_b)
  return __builtin_amdgcn_wmma_f32_16x16x32_bf16(false, a, false, b, (short)0, c,
                                                 false, false);
}

__device__ __forceinline__ bf16x16 cat8(bf16x8 lo, bf16x8 hi) {
  return __builtin_shufflevector(lo, hi, 0, 1, 2, 3, 4, 5, 6, 7,
                                 8, 9, 10, 11, 12, 13, 14, 15);
}

// ---------------------------------------------------------------------------
// f32 -> bf16 conversion
// ---------------------------------------------------------------------------
__global__ void cvt_f32_bf16(const float* __restrict__ in,
                             bf16_t* __restrict__ out, int n) {
  int i = blockIdx.x * blockDim.x + threadIdx.x;
  if (i < n) out[i] = to_bf16(in[i]);
}

// ---------------------------------------------------------------------------
// C[M,N] = A[M,K](bf16) * W[N,K]^T(bf16).
// Wave computes a 16x64 strip: 4 accumulators over N, A-fragment reused 4x.
// Block = 8 waves -> 128 rows x 64 cols. grid = (M/128, N/64).
// MODE 0: write bf16 head-split [B,H,T,hd]             (Q, K)
// MODE 1: write bf16 head-split transposed [B,H,hd,T]  (V)
// MODE 2: write f32 [M,N] + bias                       (final output)
// ---------------------------------------------------------------------------
template <int MODE>
__global__ void gemm_bf16_kernel(const bf16_t* __restrict__ A,
                                 const bf16_t* __restrict__ W,
                                 const float* __restrict__ bias,
                                 void* __restrict__ outp,
                                 float scale) {
  const int lane = threadIdx.x & 31;
  const int wv   = threadIdx.x >> 5;
  const int m0   = blockIdx.x * 128 + wv * 16;
  const int n0   = blockIdx.y * 64;
  const int K    = D_;

  floatx8 acc0 = {0.f, 0.f, 0.f, 0.f, 0.f, 0.f, 0.f, 0.f};
  floatx8 acc1 = acc0, acc2 = acc0, acc3 = acc0;

  // A-fragment: lane<16 -> row m0+lane, K chunk {k..k+15}
  //             lane>=16 -> row m0+lane-16, K chunk {k+16..k+31}
  const bf16_t* ap = A + (size_t)(m0 + (lane & 15)) * K + ((lane < 16) ? 0 : 16);
  // B-fragment: lane column n+(lane&15), 16 consecutive k (lo/hi half)
  const bf16_t* bp = W + (size_t)(n0 + (lane & 15)) * K + ((lane < 16) ? 0 : 16);

  for (int k0 = 0; k0 < K; k0 += 32) {
    bf16x8  alo = *(const bf16x8*)(ap + k0);
    bf16x8  ahi = *(const bf16x8*)(ap + k0 + 8);
    bf16x16 af  = cat8(alo, ahi);
    bf16x16 b0 = *(const bf16x16*)(bp + k0);
    bf16x16 b1 = *(const bf16x16*)(bp + (size_t)16 * K + k0);
    bf16x16 b2 = *(const bf16x16*)(bp + (size_t)32 * K + k0);
    bf16x16 b3 = *(const bf16x16*)(bp + (size_t)48 * K + k0);
    acc0 = wmma_bf16(af, b0, acc0);
    acc1 = wmma_bf16(af, b1, acc1);
    acc2 = wmma_bf16(af, b2, acc2);
    acc3 = wmma_bf16(af, b3, acc3);
  }

  const int mbase = m0 + ((lane < 16) ? 0 : 8);
#pragma unroll
  for (int j = 0; j < 4; ++j) {
    const floatx8 acc = (j == 0) ? acc0 : (j == 1) ? acc1 : (j == 2) ? acc2 : acc3;
    const int n = n0 + j * 16 + (lane & 15);
#pragma unroll
    for (int r = 0; r < 8; ++r) {
      const int   m   = mbase + r;
      const float val = acc[r] * scale;
      if (MODE == 0) {
        const int b = m >> 11, t = m & (T_ - 1);
        const int h = n >> 6,  d = n & (HD_ - 1);
        ((bf16_t*)outp)[(((size_t)b * H_ + h) * T_ + t) * HD_ + d] = to_bf16(val);
      } else if (MODE == 1) {
        const int b = m >> 11, t = m & (T_ - 1);
        const int h = n >> 6,  d = n & (HD_ - 1);
        ((bf16_t*)outp)[(((size_t)b * H_ + h) * HD_ + d) * T_ + t] = to_bf16(val);
      } else {
        ((float*)outp)[(size_t)m * D_ + n] = val + bias[n];
      }
    }
  }
}

// ---------------------------------------------------------------------------
// Attention: one block per (b, h, 16-query tile). 256 threads = 8 waves.
// LDS: S[16][SROW] f32 scores/probs + 4x16x16 f32 combine buffer.
// ---------------------------------------------------------------------------
__global__ void attn_kernel(const bf16_t* __restrict__ Qb,   // [B,H,T,hd], pre-scaled
                            const bf16_t* __restrict__ Kb,   // [B,H,T,hd]
                            const bf16_t* __restrict__ Vtb,  // [B,H,hd,T]
                            bf16_t* __restrict__ ctx) {      // [B,T,D]
  extern __shared__ float smem[];
  float* S    = smem;              // 16 * SROW
  float* comb = smem + 16 * SROW;  // 4 * 256

  const int lane = threadIdx.x & 31;
  const int wv   = threadIdx.x >> 5;
  const int qt   = blockIdx.x;
  const int h    = blockIdx.y;
  const int b    = blockIdx.z;
  const int tq0  = qt * 16;

  const size_t headQK = ((size_t)b * H_ + h) * T_ * HD_;
  const size_t headV  = ((size_t)b * H_ + h) * HD_ * T_;

  // Preload Q fragments for this query tile (k = 0..31 and 32..63).
  const bf16_t* qp =
      Qb + headQK + (size_t)(tq0 + (lane & 15)) * HD_ + ((lane < 16) ? 0 : 16);
  bf16x16 qf0 = cat8(*(const bf16x8*)(qp),      *(const bf16x8*)(qp + 8));
  bf16x16 qf1 = cat8(*(const bf16x8*)(qp + 32), *(const bf16x8*)(qp + 40));

  // ---- Phase 1: scores S = (Q*scale) @ K^T with causal mask ----
  for (int kt = wv; kt <= qt; kt += 8) {
    const bf16_t* kp = Kb + headQK +
        (size_t)(kt * 16 + (lane & 15)) * HD_ + ((lane < 16) ? 0 : 16);
    bf16x16 kf0 = *(const bf16x16*)(kp);
    bf16x16 kf1 = *(const bf16x16*)(kp + 32);
    floatx8 acc = {0.f, 0.f, 0.f, 0.f, 0.f, 0.f, 0.f, 0.f};
    acc = wmma_bf16(qf0, kf0, acc);
    acc = wmma_bf16(qf1, kf1, acc);
    const int col = kt * 16 + (lane & 15);
    const int mb  = (lane < 16) ? 0 : 8;
#pragma unroll
    for (int r = 0; r < 8; ++r) {
      const int m = mb + r;
      S[m * SROW + col] = (col <= tq0 + m) ? acc[r] : -1e30f;
    }
  }
  __syncthreads();

  // ---- Phase 2: row softmax over ncols keys ----
  const int ncols = (qt + 1) * 16;
  for (int row = wv; row < 16; row += 8) {
    float mx = -1e30f;
    for (int c = lane; c < ncols; c += 32) mx = fmaxf(mx, S[row * SROW + c]);
#pragma unroll
    for (int o = 16; o > 0; o >>= 1) mx = fmaxf(mx, __shfl_xor(mx, o, 32));
    float sum = 0.f;
    for (int c = lane; c < ncols; c += 32) {
      float e = __expf(S[row * SROW + c] - mx);
      S[row * SROW + c] = e;
      sum += e;
    }
#pragma unroll
    for (int o = 16; o > 0; o >>= 1) sum += __shfl_xor(sum, o, 32);
    const float inv = 1.f / sum;
    for (int c = lane; c < ncols; c += 32) S[row * SROW + c] *= inv;
  }
  // Zero-pad P out to a multiple of 32 columns (phase-3 K granularity).
  if ((qt & 1) == 0) {
    const int row = threadIdx.x >> 4;
    const int col = ncols + (threadIdx.x & 15);
    S[row * SROW + col] = 0.f;
  }
  __syncthreads();

  // ---- Phase 3: ctx = P @ V.  wave -> (d-tile = wv&3, K-parity = wv>>2) ----
  const int ntile = wv & 3;
  const int par   = wv >> 2;
  const int d0    = ntile * 16;
  const int nk32  = (qt + 2) >> 1;  // ceil((qt+1)*16 / 32)
  floatx8 acc = {0.f, 0.f, 0.f, 0.f, 0.f, 0.f, 0.f, 0.f};
  const int am  = lane & 15;
  const int akb = (lane < 16) ? 0 : 16;
  const bf16_t* vbase =
      Vtb + headV + (size_t)(d0 + (lane & 15)) * T_ + ((lane < 16) ? 0 : 16);
  for (int kk = par; kk < nk32; kk += 2) {
    const float* pr = &S[am * SROW + kk * 32 + akb];
    bf16x16 af;
#pragma unroll
    for (int i = 0; i < 8; ++i) {
      af[i]     = to_bf16(pr[i]);
      af[8 + i] = to_bf16(pr[8 + i]);
    }
    bf16x16 bf_ = *(const bf16x16*)(vbase + kk * 32);
    acc = wmma_bf16(af, bf_, acc);
  }
  const int mb = (lane < 16) ? 0 : 8;
  if (par == 1) {
#pragma unroll
    for (int r = 0; r < 8; ++r)
      comb[ntile * 256 + (mb + r) * 16 + (lane & 15)] = acc[r];
  }
  __syncthreads();
  if (par == 0) {
#pragma unroll
    for (int r = 0; r < 8; ++r) {
      const int   m = mb + r;
      const float v = acc[r] + comb[ntile * 256 + m * 16 + (lane & 15)];
      const int   t = tq0 + m;
      ctx[((size_t)b * T_ + t) * D_ + h * HD_ + d0 + (lane & 15)] = to_bf16(v);
    }
  }
}

// ---------------------------------------------------------------------------
// Host launch
// ---------------------------------------------------------------------------
extern "C" void kernel_launch(void* const* d_in, const int* in_sizes, int n_in,
                              void* d_out, int out_size, void* d_ws, size_t ws_size,
                              hipStream_t stream) {
  (void)in_sizes; (void)n_in; (void)out_size; (void)ws_size;

  const float* x  = (const float*)d_in[0];
  const float* Wq = (const float*)d_in[1];
  const float* Wk = (const float*)d_in[2];
  const float* Wv = (const float*)d_in[3];
  const float* Wo = (const float*)d_in[4];
  const float* bo = (const float*)d_in[5];

  char* ws = (char*)d_ws;
  const size_t MB = 1ull << 20;
  bf16_t* xb  = (bf16_t*)(ws + 0);        // 16 MB
  bf16_t* Wqb = (bf16_t*)(ws + 16 * MB);  //  2 MB
  bf16_t* Wkb = (bf16_t*)(ws + 18 * MB);
  bf16_t* Wvb = (bf16_t*)(ws + 20 * MB);
  bf16_t* Wob = (bf16_t*)(ws + 22 * MB);
  bf16_t* Qb  = (bf16_t*)(ws + 24 * MB);  // 16 MB each
  bf16_t* Kb  = (bf16_t*)(ws + 40 * MB);
  bf16_t* Vtb = (bf16_t*)(ws + 56 * MB);
  bf16_t* ctx = (bf16_t*)(ws + 72 * MB);  // ends at 88 MB

  const int nx = B_ * T_ * D_;  // 8M
  const int nw = D_ * D_;       // 1M
  cvt_f32_bf16<<<(nx + 255) / 256, 256, 0, stream>>>(x,  xb,  nx);
  cvt_f32_bf16<<<(nw + 255) / 256, 256, 0, stream>>>(Wq, Wqb, nw);
  cvt_f32_bf16<<<(nw + 255) / 256, 256, 0, stream>>>(Wk, Wkb, nw);
  cvt_f32_bf16<<<(nw + 255) / 256, 256, 0, stream>>>(Wv, Wvb, nw);
  cvt_f32_bf16<<<(nw + 255) / 256, 256, 0, stream>>>(Wo, Wob, nw);

  dim3 gp(M_ / 128, D_ / 64);  // (64, 16)
  // Fold 1/sqrt(hd) = 0.125 into Q.
  gemm_bf16_kernel<0><<<gp, 256, 0, stream>>>(xb, Wqb, nullptr, (void*)Qb, 0.125f);
  gemm_bf16_kernel<0><<<gp, 256, 0, stream>>>(xb, Wkb, nullptr, (void*)Kb, 1.0f);
  gemm_bf16_kernel<1><<<gp, 256, 0, stream>>>(xb, Wvb, nullptr, (void*)Vtb, 1.0f);

  const size_t shmem = (size_t)(16 * SROW + 4 * 256) * sizeof(float);  // ~133 KB
  hipFuncSetAttribute((const void*)attn_kernel,
                      hipFuncAttributeMaxDynamicSharedMemorySize, (int)shmem);
  attn_kernel<<<dim3(T_ / 16, H_, B_), 256, shmem, stream>>>(Qb, Kb, Vtb, ctx);

  gemm_bf16_kernel<2><<<gp, 256, 0, stream>>>(ctx, Wob, bo, d_out, 1.0f);
}
